// Denoiser_16887811408713
// MI455X (gfx1250) — matrix operand order repaired and checked by reading, not compiled, not run on hardware
//
#include <hip/hip_runtime.h>

// ---------------------------------------------------------------------------
// Problem dims (fixed by the reference)
// ---------------------------------------------------------------------------
#define B_   2
#define T_   1024
#define L_   6
#define E_   512
#define D_   768
#define H_   12
#define DH_  64
#define NTOK (B_ * T_)   // 2048

// ---------------------------------------------------------------------------
// CDNA5 async global->LDS support (probe-guarded; falls back to load+ds_store)
// Builtin signature (from hipcc diagnostic): (int4 as(1)*, int4 as(3)*, imm, imm)
// ---------------------------------------------------------------------------
#if defined(__AMDGCN__) && defined(__has_builtin)
#if __has_builtin(__builtin_amdgcn_global_load_async_to_lds_b128)
#define ASYNC_LDS 1
#endif
#endif
#ifndef ASYNC_LDS
#define ASYNC_LDS 0
#endif

#if ASYNC_LDS
typedef int i4g __attribute__((vector_size(16)));
typedef __attribute__((address_space(1))) i4g* gi4p;
typedef __attribute__((address_space(3))) i4g* li4p;
#define ASYNC_CP_B128(gp, lp) \
    __builtin_amdgcn_global_load_async_to_lds_b128((gi4p)(gp), (li4p)(lp), 0, 0)
#if __has_builtin(__builtin_amdgcn_s_wait_asynccnt)
#define WAIT_ASYNC() __builtin_amdgcn_s_wait_asynccnt(0)
#else
#define WAIT_ASYNC() asm volatile("s_wait_asynccnt 0x0" ::: "memory")
#endif
#endif

// ---------------------------------------------------------------------------
// Vector types for WMMA operands (wave32, CDNA5 layouts per CDNA5 ISA)
// ---------------------------------------------------------------------------
typedef _Float16     h16v __attribute__((ext_vector_type(16)));
typedef float        f8v  __attribute__((ext_vector_type(8)));
typedef unsigned int u4v  __attribute__((ext_vector_type(4)));

union H16U { h16v h; u4v u[2]; };
union H8U  { u4v u; _Float16 h[8]; };

__device__ __forceinline__ h16v pack16(u4v lo, u4v hi) {
    H16U t; t.u[0] = lo; t.u[1] = hi; return t.h;
}
__device__ __forceinline__ f8v zero_f8() {
    f8v z = {0.f, 0.f, 0.f, 0.f, 0.f, 0.f, 0.f, 0.f};
    return z;
}
__device__ __forceinline__ float gelu_tanh(float x) {
    float x3 = x * x * x;
    return 0.5f * x * (1.0f + tanhf(0.7978845608028654f * (x + 0.044715f * x3)));
}
__device__ __forceinline__ float gelu_exact(float x) {
    return 0.5f * x * (1.0f + erff(x * 0.7071067811865475f));
}

// ---------------------------------------------------------------------------
// Workspace layout (bytes, all 256-aligned)
// ---------------------------------------------------------------------------
static constexpr size_t OFF_X     = 0;                                     // f32 [2048,512]
static constexpr size_t OFF_X16   = OFF_X     + (size_t)NTOK * E_ * 4;     // f16 [2048,512]
static constexpr size_t OFF_H     = OFF_X16   + (size_t)NTOK * E_ * 2;     // f32 [2048,768]
static constexpr size_t OFF_A16   = OFF_H     + (size_t)NTOK * D_ * 4;     // f16 [2048,3072]
static constexpr size_t OFF_M16   = OFF_A16   + (size_t)NTOK * 4 * D_ * 2; // f16 [2048,3072]
static constexpr size_t OFF_C16   = OFF_M16   + (size_t)NTOK * 4 * D_ * 2; // f16 [2048,768]
static constexpr size_t OFF_QKV16 = OFF_C16   + (size_t)NTOK * D_ * 2;     // f16 [2048,2304]
static constexpr size_t OFF_W16   = OFF_QKV16 + (size_t)NTOK * 3 * D_ * 2; // f16 weight scratch
static constexpr size_t OFF_FEMB  = OFF_W16   + (size_t)(4 * D_) * D_ * 2; // f32 [2,512]
static constexpr size_t OFF_TEMB  = OFF_FEMB  + 4096;                      // f32 [2,512]

// ---------------------------------------------------------------------------
// Prologue kernels
// ---------------------------------------------------------------------------
__global__ void femb_kernel(const float* __restrict__ ts, float* __restrict__ femb) {
    int idx = blockIdx.x * 256 + threadIdx.x;
    if (idx >= B_ * E_) return;
    int b = idx / E_, e = idx % E_;
    const int half = E_ / 2;
    int i = (e < half) ? e : e - half;
    float f = __expf((float)i * (-9.210340371976184f / (float)(half - 1)));
    float a = ts[b] * f;
    femb[idx] = (e < half) ? sinf(a) : cosf(a);
}

__global__ void temb_kernel(const float* __restrict__ femb, const float* __restrict__ tw,
                            const float* __restrict__ tb, float* __restrict__ temb) {
    int idx = blockIdx.x * 256 + threadIdx.x;
    if (idx >= B_ * E_) return;
    int b = idx / E_, n = idx % E_;
    const float* fr = femb + b * E_;
    float acc = tb[n];
    for (int k = 0; k < E_; ++k) acc += fr[k] * tw[k * E_ + n];
    temb[idx] = gelu_exact(acc);
}

__global__ void addx_kernel(const float* __restrict__ inp, const float* __restrict__ wpe,
                            const float* __restrict__ temb,
                            float* __restrict__ X, _Float16* __restrict__ X16) {
    int idx = blockIdx.x * 256 + threadIdx.x;
    if (idx >= NTOK * E_) return;
    int e = idx % E_;
    int t = (idx / E_) % T_;
    int b = idx / (T_ * E_);
    float v = inp[idx] + wpe[t * E_ + e] + temb[b * E_ + e];
    X[idx]   = v;
    X16[idx] = (_Float16)v;
}

// W [K,N] f32 row-major -> Wt [N,K] f16 row-major via 32x32 LDS tile transpose.
// Grid: (N/32, K/32), block 256 (8 rows of 32 lanes, 4 iterations).
__global__ __launch_bounds__(256) void wconv_kernel(const float* __restrict__ W,
                                                    _Float16* __restrict__ Wt,
                                                    int K, int N) {
    __shared__ _Float16 tile[32][33];
    int nb = blockIdx.x * 32, kb = blockIdx.y * 32;
    int tx = threadIdx.x & 31, ty = threadIdx.x >> 5;
    #pragma unroll
    for (int i = 0; i < 4; ++i) {
        int k = ty + i * 8;
        tile[k][tx] = (_Float16)W[(size_t)(kb + k) * N + nb + tx];
    }
    __syncthreads();
    #pragma unroll
    for (int i = 0; i < 4; ++i) {
        int n = ty + i * 8;
        Wt[(size_t)(nb + n) * K + kb + tx] = tile[tx][n];
    }
}

// ---------------------------------------------------------------------------
// LayerNorm: one block per row, emits f16 (next GEMM's A operand)
// ---------------------------------------------------------------------------
__global__ __launch_bounds__(256) void ln_kernel(const float* __restrict__ x,
                                                 const float* __restrict__ g,
                                                 const float* __restrict__ b,
                                                 _Float16* __restrict__ out16, int Dn) {
    int row = blockIdx.x;
    const float* xr = x + (size_t)row * Dn;
    float s = 0.f, s2 = 0.f;
    for (int i = threadIdx.x; i < Dn; i += 256) { float v = xr[i]; s += v; s2 += v * v; }
    #pragma unroll
    for (int off = 16; off > 0; off >>= 1) { s += __shfl_xor(s, off); s2 += __shfl_xor(s2, off); }
    __shared__ float sh[2][8];
    int wave = threadIdx.x >> 5, lane = threadIdx.x & 31;
    if (lane == 0) { sh[0][wave] = s; sh[1][wave] = s2; }
    __syncthreads();
    if (threadIdx.x == 0) {
        float a = 0.f, c = 0.f;
        for (int i = 0; i < 8; ++i) { a += sh[0][i]; c += sh[1][i]; }
        sh[0][0] = a; sh[1][0] = c;
    }
    __syncthreads();
    float mean = sh[0][0] / (float)Dn;
    float var  = sh[1][0] / (float)Dn - mean * mean;
    float rs   = rsqrtf(var + 1e-5f);
    for (int i = threadIdx.x; i < Dn; i += 256)
        out16[(size_t)row * Dn + i] = (_Float16)((xr[i] - mean) * rs * g[i] + b[i]);
}

// ---------------------------------------------------------------------------
// WMMA GEMM: out = act(A[M,K] @ Wt[N,K]^T + bias) (+res), templated epilogue.
// Block 256 thr = 8 waves (4x2), block tile 128x128, wave tile 32x64, BK=32.
// Async path: double-buffered LDS tiles filled with global_load_async_to_lds_b128,
// synchronized with s_wait_asynccnt + barrier; WMMAs overlap the next prefetch.
// ---------------------------------------------------------------------------
#define BK 32
template <int ACT, bool RES, bool OUT32, bool OUT16>
__global__ __launch_bounds__(256) void gemm_wmma_kernel(
    const _Float16* __restrict__ A, const _Float16* __restrict__ Wt,
    const float* __restrict__ bias, const float* __restrict__ res,
    float* __restrict__ out, _Float16* __restrict__ out16,
    int M, int N, int K) {
    constexpr int LDT = 40;  // halves per LDS row (pad 32 -> 40, keeps 16B align)
    const int tid  = threadIdx.x;
    const int wave = tid >> 5, lane = tid & 31;
    const int wm   = wave >> 1, wn = wave & 1;
    const int bm0  = blockIdx.y * 128, bn0 = blockIdx.x * 128;
    const int lr   = lane & 15, hs = lane >> 4;

    // per-thread staging coords: 2 x 16B segments for A and for B
    const int r0 = (tid * 2) >> 2, s0 = (tid * 2) & 3;
    const int r1 = (tid * 2 + 1) >> 2, s1 = (tid * 2 + 1) & 3;

    f8v acc[2][4];
    #pragma unroll
    for (int mi = 0; mi < 2; ++mi)
        #pragma unroll
        for (int ni = 0; ni < 4; ++ni) acc[mi][ni] = zero_f8();

#if ASYNC_LDS
    __shared__ __align__(16) _Float16 sA[2][128 * LDT];
    __shared__ __align__(16) _Float16 sB[2][128 * LDT];
    auto stage = [&](int k0, int buf) {
        ASYNC_CP_B128(A  + (size_t)(bm0 + r0) * K + k0 + s0 * 8, &sA[buf][r0 * LDT + s0 * 8]);
        ASYNC_CP_B128(A  + (size_t)(bm0 + r1) * K + k0 + s1 * 8, &sA[buf][r1 * LDT + s1 * 8]);
        ASYNC_CP_B128(Wt + (size_t)(bn0 + r0) * K + k0 + s0 * 8, &sB[buf][r0 * LDT + s0 * 8]);
        ASYNC_CP_B128(Wt + (size_t)(bn0 + r1) * K + k0 + s1 * 8, &sB[buf][r1 * LDT + s1 * 8]);
    };
    stage(0, 0);
    WAIT_ASYNC();
    __syncthreads();
    int buf = 0;
    for (int k0 = 0; k0 < K; k0 += BK) {
        if (k0 + BK < K) stage(k0 + BK, buf ^ 1);   // prefetch next tile
        h16v af[2], bf[4];
        #pragma unroll
        for (int mi = 0; mi < 2; ++mi) {
            int r = wm * 32 + mi * 16 + lr;
            u4v lo = *reinterpret_cast<const u4v*>(&sA[buf][r * LDT + hs * 8]);
            u4v hi = *reinterpret_cast<const u4v*>(&sA[buf][r * LDT + 16 + hs * 8]);
            af[mi] = pack16(lo, hi);
        }
        #pragma unroll
        for (int ni = 0; ni < 4; ++ni) {
            int r = wn * 64 + ni * 16 + lr;
            u4v lo = *reinterpret_cast<const u4v*>(&sB[buf][r * LDT + hs * 8]);
            u4v hi = *reinterpret_cast<const u4v*>(&sB[buf][r * LDT + 16 + hs * 8]);
            bf[ni] = pack16(lo, hi);
        }
        #pragma unroll
        for (int mi = 0; mi < 2; ++mi)
            #pragma unroll
            for (int ni = 0; ni < 4; ++ni)
                acc[mi][ni] = __builtin_amdgcn_wmma_f32_16x16x32_f16(
                    false, af[mi], false, bf[ni], (short)0, acc[mi][ni], false, false);
        WAIT_ASYNC();      // next tile landed in LDS
        __syncthreads();   // all waves done reading current tile
        buf ^= 1;
    }
#else
    __shared__ __align__(16) _Float16 sA[128 * LDT];
    __shared__ __align__(16) _Float16 sB[128 * LDT];
    for (int k0 = 0; k0 < K; k0 += BK) {
        *reinterpret_cast<u4v*>(&sA[r0 * LDT + s0 * 8]) =
            *reinterpret_cast<const u4v*>(A + (size_t)(bm0 + r0) * K + k0 + s0 * 8);
        *reinterpret_cast<u4v*>(&sA[r1 * LDT + s1 * 8]) =
            *reinterpret_cast<const u4v*>(A + (size_t)(bm0 + r1) * K + k0 + s1 * 8);
        *reinterpret_cast<u4v*>(&sB[r0 * LDT + s0 * 8]) =
            *reinterpret_cast<const u4v*>(Wt + (size_t)(bn0 + r0) * K + k0 + s0 * 8);
        *reinterpret_cast<u4v*>(&sB[r1 * LDT + s1 * 8]) =
            *reinterpret_cast<const u4v*>(Wt + (size_t)(bn0 + r1) * K + k0 + s1 * 8);
        __syncthreads();
        h16v af[2], bf[4];
        #pragma unroll
        for (int mi = 0; mi < 2; ++mi) {
            int r = wm * 32 + mi * 16 + lr;
            u4v lo = *reinterpret_cast<const u4v*>(&sA[r * LDT + hs * 8]);
            u4v hi = *reinterpret_cast<const u4v*>(&sA[r * LDT + 16 + hs * 8]);
            af[mi] = pack16(lo, hi);
        }
        #pragma unroll
        for (int ni = 0; ni < 4; ++ni) {
            int r = wn * 64 + ni * 16 + lr;
            u4v lo = *reinterpret_cast<const u4v*>(&sB[r * LDT + hs * 8]);
            u4v hi = *reinterpret_cast<const u4v*>(&sB[r * LDT + 16 + hs * 8]);
            bf[ni] = pack16(lo, hi);
        }
        #pragma unroll
        for (int mi = 0; mi < 2; ++mi)
            #pragma unroll
            for (int ni = 0; ni < 4; ++ni)
                acc[mi][ni] = __builtin_amdgcn_wmma_f32_16x16x32_f16(
                    false, af[mi], false, bf[ni], (short)0, acc[mi][ni], false, false);
        __syncthreads();
    }
#endif

    // epilogue: C layout = lane col l%16, vgpr r -> row r + 8*(l/16)
    #pragma unroll
    for (int mi = 0; mi < 2; ++mi) {
        #pragma unroll
        for (int ni = 0; ni < 4; ++ni) {
            int col  = bn0 + wn * 64 + ni * 16 + lr;
            float bv = bias[col];
            #pragma unroll
            for (int r = 0; r < 8; ++r) {
                int row = bm0 + wm * 32 + mi * 16 + r + 8 * hs;
                float v = acc[mi][ni][r] + bv;
                if constexpr (ACT == 1) v = gelu_tanh(v);
                size_t o = (size_t)row * N + col;
                if constexpr (RES)   v += res[o];
                if constexpr (OUT32) out[o] = v;
                if constexpr (OUT16) out16[o] = (_Float16)v;
            }
        }
    }
}

// ---------------------------------------------------------------------------
// Flash attention (no causal mask). Block = 128 thr (4 waves), grid (T/64, B*H).
// ---------------------------------------------------------------------------
__global__ __launch_bounds__(128) void attn_kernel(
    const _Float16* __restrict__ qkv, _Float16* __restrict__ y16) {
    const int RS = 3 * D_;  // 2304 halves per token row
    const int wave = threadIdx.x >> 5, lane = threadIdx.x & 31;
    const int lr = lane & 15, hs = lane >> 4;
    const int bh = blockIdx.y, b = bh / H_, h = bh % H_;
    const int q0 = blockIdx.x * 64 + wave * 16;

    const _Float16* qp = qkv + (size_t)b * T_ * RS + h * DH_;
    const _Float16* kp = qp + D_;
    const _Float16* vp = qp + 2 * D_;

    h16v qf[2];
    #pragma unroll
    for (int c = 0; c < 2; ++c) {
        const _Float16* base = qp + (size_t)(q0 + lr) * RS + c * 32;
        u4v lo = *reinterpret_cast<const u4v*>(base + hs * 8);
        u4v hi = *reinterpret_cast<const u4v*>(base + 16 + hs * 8);
        qf[c] = pack16(lo, hi);
    }

    f8v o[4];
    #pragma unroll
    for (int nt = 0; nt < 4; ++nt) o[nt] = zero_f8();
    float mi_[8], li_[8];
    #pragma unroll
    for (int r = 0; r < 8; ++r) { mi_[r] = -1e30f; li_[r] = 0.f; }

    __shared__ __align__(16) _Float16 sP[4][16 * 32];
    __shared__ __align__(16) _Float16 sV[64 * 32];

    for (int kc = 0; kc < T_; kc += 32) {
        __syncthreads();
        // stage transposed V chunk: coalesced b128 global loads, b16 LDS scatter.
        // seg = key*8 + hd_octet; each thread moves 8 head-dim elems of one key.
        #pragma unroll
        for (int it = 0; it < 2; ++it) {
            int seg = threadIdx.x + it * 128;   // 0..255
            int key = seg >> 3, hb = (seg & 7) * 8;
            H8U t;
            t.u = *reinterpret_cast<const u4v*>(vp + (size_t)(kc + key) * RS + hb);
            #pragma unroll
            for (int j = 0; j < 8; ++j)
                sV[(hb + j) * 32 + key] = t.h[j];
        }
        f8v st[2];
        #pragma unroll
        for (int ks = 0; ks < 2; ++ks) {
            const _Float16* base = kp + (size_t)(kc + ks * 16 + lr) * RS;
            u4v lo0 = *reinterpret_cast<const u4v*>(base + hs * 8);
            u4v hi0 = *reinterpret_cast<const u4v*>(base + 16 + hs * 8);
            u4v lo1 = *reinterpret_cast<const u4v*>(base + 32 + hs * 8);
            u4v hi1 = *reinterpret_cast<const u4v*>(base + 48 + hs * 8);
            h16v kf0 = pack16(lo0, hi0);
            h16v kf1 = pack16(lo1, hi1);
            f8v sa = zero_f8();
            sa = __builtin_amdgcn_wmma_f32_16x16x32_f16(false, qf[0], false, kf0, (short)0, sa, false, false);
            sa = __builtin_amdgcn_wmma_f32_16x16x32_f16(false, qf[1], false, kf1, (short)0, sa, false, false);
            st[ks] = sa * 0.125f;
        }
        float tmax[8];
        #pragma unroll
        for (int r = 0; r < 8; ++r) tmax[r] = fmaxf(st[0][r], st[1][r]);
        #pragma unroll
        for (int r = 0; r < 8; ++r)
            #pragma unroll
            for (int off = 1; off < 16; off <<= 1)
                tmax[r] = fmaxf(tmax[r], __shfl_xor(tmax[r], off));
        float alpha[8], tsum[8];
        #pragma unroll
        for (int r = 0; r < 8; ++r) {
            float mnew = fmaxf(mi_[r], tmax[r]);
            alpha[r] = __expf(mi_[r] - mnew);
            mi_[r] = mnew; tsum[r] = 0.f;
        }
        #pragma unroll
        for (int ks = 0; ks < 2; ++ks)
            #pragma unroll
            for (int r = 0; r < 8; ++r) {
                float p = __expf(st[ks][r] - mi_[r]);
                st[ks][r] = p; tsum[r] += p;
            }
        #pragma unroll
        for (int r = 0; r < 8; ++r) {
            #pragma unroll
            for (int off = 1; off < 16; off <<= 1) tsum[r] += __shfl_xor(tsum[r], off);
            li_[r] = li_[r] * alpha[r] + tsum[r];
        }
        #pragma unroll
        for (int nt = 0; nt < 4; ++nt)
            #pragma unroll
            for (int r = 0; r < 8; ++r) o[nt][r] *= alpha[r];
        #pragma unroll
        for (int ks = 0; ks < 2; ++ks)
            #pragma unroll
            for (int r = 0; r < 8; ++r)
                sP[wave][(r + 8 * hs) * 32 + ks * 16 + lr] = (_Float16)st[ks][r];
        __syncthreads();
        u4v plo = *reinterpret_cast<const u4v*>(&sP[wave][lr * 32 + hs * 8]);
        u4v phi = *reinterpret_cast<const u4v*>(&sP[wave][lr * 32 + 16 + hs * 8]);
        h16v pf = pack16(plo, phi);
        #pragma unroll
        for (int nt = 0; nt < 4; ++nt) {
            u4v lo = *reinterpret_cast<const u4v*>(&sV[(nt * 16 + lr) * 32 + hs * 8]);
            u4v hi = *reinterpret_cast<const u4v*>(&sV[(nt * 16 + lr) * 32 + 16 + hs * 8]);
            h16v vf = pack16(lo, hi);
            o[nt] = __builtin_amdgcn_wmma_f32_16x16x32_f16(false, pf, false, vf, (short)0, o[nt], false, false);
        }
    }
    #pragma unroll
    for (int nt = 0; nt < 4; ++nt)
        #pragma unroll
        for (int r = 0; r < 8; ++r) {
            int row = q0 + r + 8 * hs;
            int col = nt * 16 + lr;
            float val = o[nt][r] / li_[r];
            y16[(size_t)(b * T_ + row) * D_ + h * DH_ + col] = (_Float16)val;
        }
}

// ---------------------------------------------------------------------------
// Host orchestration
// ---------------------------------------------------------------------------
template <int ACT, bool RES, bool OUT32, bool OUT16>
static void launch_gemm(const _Float16* A, const _Float16* Wt, const float* bias,
                        const float* res, float* out, _Float16* out16,
                        int M, int N, int K, hipStream_t stream) {
    dim3 g(N / 128, M / 128);
    gemm_wmma_kernel<ACT, RES, OUT32, OUT16><<<g, 256, 0, stream>>>(
        A, Wt, bias, res, out, out16, M, N, K);
}

extern "C" void kernel_launch(void* const* d_in, const int* in_sizes, int n_in,
                              void* d_out, int out_size, void* d_ws, size_t ws_size,
                              hipStream_t stream) {
    (void)in_sizes; (void)n_in; (void)out_size; (void)ws_size;

    const float* inp     = (const float*)d_in[0];
    const float* ts      = (const float*)d_in[1];
    const float* wpe     = (const float*)d_in[2];
    const float* time_w  = (const float*)d_in[3];
    const float* time_b  = (const float*)d_in[4];
    const float* up_w    = (const float*)d_in[5];
    const float* up_b    = (const float*)d_in[6];
    const float* ln1_g   = (const float*)d_in[7];
    const float* ln1_b   = (const float*)d_in[8];
    const float* attn_w  = (const float*)d_in[9];
    const float* attn_b  = (const float*)d_in[10];
    const float* attnp_w = (const float*)d_in[11];
    const float* attnp_b = (const float*)d_in[12];
    const float* ln2_g   = (const float*)d_in[13];
    const float* ln2_b   = (const float*)d_in[14];
    const float* fc_w    = (const float*)d_in[15];
    const float* fc_b    = (const float*)d_in[16];
    const float* fcp_w   = (const float*)d_in[17];
    const float* fcp_b   = (const float*)d_in[18];
    const float* down_w  = (const float*)d_in[19];
    const float* down_b  = (const float*)d_in[20];
    const float* lnf_g   = (const float*)d_in[21];
    const float* lnf_b   = (const float*)d_in[22];
    const float* small_w = (const float*)d_in[23];
    const float* small_b = (const float*)d_in[24];

    char* ws = (char*)d_ws;
    float*    X     = (float*)(ws + OFF_X);
    _Float16* X16   = (_Float16*)(ws + OFF_X16);
    float*    Hb    = (float*)(ws + OFF_H);
    _Float16* A16   = (_Float16*)(ws + OFF_A16);
    _Float16* M16   = (_Float16*)(ws + OFF_M16);
    _Float16* C16   = (_Float16*)(ws + OFF_C16);
    _Float16* QKV16 = (_Float16*)(ws + OFF_QKV16);
    _Float16* W16   = (_Float16*)(ws + OFF_W16);
    float*    FEMB  = (float*)(ws + OFF_FEMB);
    float*    TEMB  = (float*)(ws + OFF_TEMB);

    auto conv = [&](const float* W, int K, int N) {
        wconv_kernel<<<dim3(N / 32, K / 32), 256, 0, stream>>>(W, W16, K, N);
    };

    // prologue
    femb_kernel<<<(B_ * E_ + 255) / 256, 256, 0, stream>>>(ts, FEMB);
    temb_kernel<<<(B_ * E_ + 255) / 256, 256, 0, stream>>>(FEMB, time_w, time_b, TEMB);
    addx_kernel<<<(NTOK * E_ + 255) / 256, 256, 0, stream>>>(inp, wpe, TEMB, X, X16);

    for (int l = 0; l < L_; ++l) {
        // h = x @ up_w + up_b
        conv(up_w + (size_t)l * E_ * D_, E_, D_);
        launch_gemm<0, false, true, false>(X16, W16, up_b + l * D_, nullptr, Hb, nullptr,
                                           NTOK, D_, E_, stream);
        // qkv = ln1(h) @ attn_w + attn_b  (f16 only, feeds attention)
        ln_kernel<<<NTOK, 256, 0, stream>>>(Hb, ln1_g + l * D_, ln1_b + l * D_, A16, D_);
        conv(attn_w + (size_t)l * D_ * 3 * D_, D_, 3 * D_);
        launch_gemm<0, false, false, true>(A16, W16, attn_b + l * 3 * D_, nullptr, nullptr,
                                           QKV16, NTOK, 3 * D_, D_, stream);
        // y = softmax(qk/8) v  -> f16 into A16
        attn_kernel<<<dim3(T_ / 64, B_ * H_), 128, 0, stream>>>(QKV16, A16);
        // h += y @ attnp_w + attnp_b
        conv(attnp_w + (size_t)l * D_ * D_, D_, D_);
        launch_gemm<0, true, true, false>(A16, W16, attnp_b + l * D_, Hb, Hb, nullptr,
                                          NTOK, D_, D_, stream);
        // m = gelu_tanh(ln2(h) @ fc_w + fc_b)
        ln_kernel<<<NTOK, 256, 0, stream>>>(Hb, ln2_g + l * D_, ln2_b + l * D_, A16, D_);
        conv(fc_w + (size_t)l * D_ * 4 * D_, D_, 4 * D_);
        launch_gemm<1, false, false, true>(A16, W16, fc_b + l * 4 * D_, nullptr, nullptr,
                                           M16, NTOK, 4 * D_, D_, stream);
        // h += m @ fcp_w + fcp_b  (also emit h as f16)
        conv(fcp_w + (size_t)l * 4 * D_ * D_, 4 * D_, D_);
        launch_gemm<0, true, true, true>(M16, W16, fcp_b + l * D_, Hb, Hb, C16,
                                         NTOK, D_, 4 * D_, stream);
        // x += h @ down_w + down_b (also refresh f16 x for next layer)
        conv(down_w + (size_t)l * D_ * E_, D_, E_);
        launch_gemm<0, true, true, true>(C16, W16, down_b + l * E_, X, X, X16,
                                         NTOK, E_, D_, stream);
    }
    // out = lnf(x) @ small_w + small_b
    ln_kernel<<<NTOK, 256, 0, stream>>>(X, lnf_g, lnf_b, A16, E_);
    conv(small_w, E_, E_);
    launch_gemm<0, false, true, false>(A16, W16, small_b, nullptr, (float*)d_out, nullptr,
                                       NTOK, E_, E_, stream);
}
